// CKConv_78580721647672
// MI455X (gfx1250) — matrix operand order repaired and checked by reading, not compiled, not run on hardware
//
#include <hip/hip_runtime.h>
#include <hip/hip_bf16.h>

typedef __bf16  bf16_t;
typedef __bf16  v16bf __attribute__((ext_vector_type(16)));
typedef __bf16  v8bf  __attribute__((ext_vector_type(8)));
typedef float   v8f   __attribute__((ext_vector_type(8)));

#define B_    32
#define CIN_  32
#define COUT_ 32
#define T_    2048
#define K_    2049
#define DK_   32
#define GD    2080   // tap-table delay stride (elements): 4160 B rows, every 16B chunk aligned
#define XRS   2208   // reversed-x global row length (2048 data + 160 zero pad), 4416 B rows
#define XS    2112   // LDS staging capacity per buffer (bf16)
#define TBLK  128    // time steps per workgroup (8 waves x 16)

#if defined(__gfx1250__) && __has_builtin(__builtin_amdgcn_global_load_async_to_lds_b128) && \
    __has_builtin(__builtin_amdgcn_s_wait_asynccnt)
#define HAVE_ASYNC_LDS 1
typedef int v4i_gv __attribute__((vector_size(16)));
typedef __attribute__((address_space(1))) v4i_gv* async_gptr_t;   // global source
typedef __attribute__((address_space(3))) v4i_gv* async_lptr_t;   // LDS destination
#else
#define HAVE_ASYNC_LDS 0
#endif

// ---------------------------------------------------------------------------
// Kernel A: SIREN MLP -> delay-ordered bf16 conv taps
//   gbf[p*GD + d] = bf16( kern[o][i][2048 - d] ),  p = o*CIN + i
// ---------------------------------------------------------------------------
__global__ __launch_bounds__(256) void ck_gen_taps(
    const float* __restrict__ rel_pos,
    const float* __restrict__ w1, const float* __restrict__ b1,
    const float* __restrict__ om1p,
    const float* __restrict__ w2, const float* __restrict__ b2,
    const float* __restrict__ om2p,
    const float* __restrict__ w3, const float* __restrict__ b3,
    bf16_t* __restrict__ gbf)
{
    int k = blockIdx.x * blockDim.x + threadIdx.x;
    if (k >= K_) return;
    const float pos = rel_pos[k];
    const float om1 = om1p[0], om2 = om2p[0];

    float h1[DK_], h2[DK_];
    #pragma unroll
    for (int j = 0; j < DK_; ++j)
        h1[j] = __sinf(om1 * (w1[j] * pos + b1[j]));

    #pragma unroll
    for (int j = 0; j < DK_; ++j) {
        float a = b2[j];
        #pragma unroll
        for (int m = 0; m < DK_; ++m) a += w2[j*DK_ + m] * h1[m];
        h2[j] = __sinf(om2 * a);
    }

    const int d = 2048 - k;
    for (int p = 0; p < COUT_*CIN_; ++p) {
        float a = b3[p];
        #pragma unroll
        for (int m = 0; m < DK_; ++m) a += w3[p*DK_ + m] * h2[m];
        gbf[p*GD + d] = (bf16_t)a;
    }
}

// ---------------------------------------------------------------------------
// Kernel A2: reversed, zero-padded bf16 image of x (one-time, enables
// contiguous aligned LDS staging in the conv kernel).
//   xrev[row][q] = bf16(x[row][2047-q]) for q<2048, 0 for the causal pad tail
// ---------------------------------------------------------------------------
__global__ __launch_bounds__(256) void ck_rev_x(
    const float* __restrict__ x, bf16_t* __restrict__ xrev)
{
    const int row = blockIdx.x;                 // b*CIN + i
    const float* src = x + (size_t)row * T_;
    bf16_t* dst = xrev + (size_t)row * XRS;
    for (int q = threadIdx.x; q < XRS; q += 256)
        dst[q] = (q < T_) ? (bf16_t)src[T_ - 1 - q] : (bf16_t)0.0f;
}

// ---------------------------------------------------------------------------
// Kernel B: causal conv as Toeplitz GEMM on bf16 WMMA, f32 accumulation.
//   out[b,o,t] = bias[o] + sum_i sum_d gbf[o,i,d] * x[b,i,t-d]
// 256 thr / 8 waves; block covers one batch x 32 o x 128 t. Wave w owns
// t_base = t0+16w and two 16x16 tiles (both COUT halves share one B frag).
// x slab per input channel is staged into LDS by async LDS-DMA (ASYNCcnt),
// double-buffered: channel i+1 streams in while WMMAs consume channel i.
// ---------------------------------------------------------------------------
__global__ __launch_bounds__(256) void ck_conv_wmma(
    const bf16_t* __restrict__ xrev,  // [B*CIN][XRS]
    const bf16_t* __restrict__ gbf,   // [COUT*CIN][GD]
    const float*  __restrict__ bias,  // [COUT]
    float*        __restrict__ out)   // [B][COUT][T]
{
    __shared__ __align__(16) unsigned short xbuf[2][XS];

    const int t0   = blockIdx.x * TBLK;
    const int b    = blockIdx.y;
    const int tid  = threadIdx.x;
    const int wave = tid >> 5;                // = t_sub, 0..7
    const int lane = tid & 31;
    const int half = lane >> 4;               // 0/1
    const int ln16 = lane & 15;
    const int t_base = t0 + wave * 16;

    v8f acc0 = {};                            // o in [0,16)
    v8f acc1 = {};                            // o in [16,32)

    const int dmax   = t_base + 15;           // largest delay this wave needs
    const int q0     = 1920 - t0;             // offset into reversed image
    const int nchunk = (t0 + 176) >> 3;       // 16B chunks actually consumed

    auto stage = [&](int i, int buf) {
        const bf16_t* src = xrev + ((size_t)(b*CIN_ + i) * XRS + q0);
        for (int c = tid; c < nchunk; c += 256) {
#if HAVE_ASYNC_LDS
            __builtin_amdgcn_global_load_async_to_lds_b128(
                (async_gptr_t)(src + 8*c),
                (async_lptr_t)&xbuf[buf][8*c],
                0, 0);
#else
            ((uint4*)xbuf[buf])[c] = *(const uint4*)(src + 8*c);
#endif
        }
    };

    stage(0, 0);

    for (int i = 0; i < CIN_; ++i) {
#if HAVE_ASYNC_LDS
        __builtin_amdgcn_s_wait_asynccnt(0);
#endif
        __syncthreads();                       // staged slab i visible to all
        if (i + 1 < CIN_) stage(i + 1, (i + 1) & 1);   // prefetch next channel

        const bf16_t* xr = reinterpret_cast<const bf16_t*>(xbuf[i & 1]);
        const bf16_t* grow0 = gbf + ((0  + ln16)*CIN_ + i) * (long)GD;
        const bf16_t* grow1 = gbf + ((16 + ln16)*CIN_ + i) * (long)GD;
        const int bbase = 8*half + (127 - 16*wave - ln16);   // >= 0 at d0=0

        #pragma unroll 2
        for (int d0 = 0; d0 <= dmax; d0 += 32) {
            // A fragments: 16x32 bf16 taps, two aligned 16B chunks per lane
            // (layout: j<8 -> K=8*half+j ; j>=8 -> K=16+8*half+(j-8))
            v8bf a0lo = *reinterpret_cast<const v8bf*>(grow0 + d0 +      8*half);
            v8bf a0hi = *reinterpret_cast<const v8bf*>(grow0 + d0 + 16 + 8*half);
            v8bf a1lo = *reinterpret_cast<const v8bf*>(grow1 + d0 +      8*half);
            v8bf a1hi = *reinterpret_cast<const v8bf*>(grow1 + d0 + 16 + 8*half);
            v16bf A0, A1, Bf;
            #pragma unroll
            for (int j = 0; j < 8; ++j) {
                A0[j] = a0lo[j]; A0[j+8] = a0hi[j];
                A1[j] = a1lo[j]; A1[j+8] = a1hi[j];
            }
            // B fragment: Toeplitz slab; lane N=ln16, j<8 -> K=8*half+j ;
            // j>=8 -> K=16+8*half+(j-8), in the reversed LDS image
            const int bb = d0 + bbase;
            #pragma unroll
            for (int j = 0; j < 16; ++j) Bf[j] = xr[bb + j + (j >= 8 ? 8 : 0)];

            acc0 = __builtin_amdgcn_wmma_f32_16x16x32_bf16(
                       false, A0, false, Bf, (short)0, acc0, false, false);
            acc1 = __builtin_amdgcn_wmma_f32_16x16x32_bf16(
                       false, A1, false, Bf, (short)0, acc1, false, false);
        }
    }

    // D layout: VGPR r -> M = r + 8*half (o), N = ln16 (t): coalesced stores
    const int t = t_base + ln16;
    #pragma unroll
    for (int r = 0; r < 8; ++r) {
        const int o0 = r + 8*half;
        const int o1 = 16 + r + 8*half;
        out[(b*COUT_ + o0)*T_ + t] = acc0[r] + bias[o0];
        out[(b*COUT_ + o1)*T_ + t] = acc1[r] + bias[o1];
    }
}

// ---------------------------------------------------------------------------
extern "C" void kernel_launch(void* const* d_in, const int* in_sizes, int n_in,
                              void* d_out, int out_size, void* d_ws, size_t ws_size,
                              hipStream_t stream)
{
    const float* x    = (const float*)d_in[0];
    const float* rel  = (const float*)d_in[1];
    const float* w1   = (const float*)d_in[2];
    const float* b1   = (const float*)d_in[3];
    const float* om1  = (const float*)d_in[4];
    const float* w2   = (const float*)d_in[5];
    const float* b2   = (const float*)d_in[6];
    const float* om2  = (const float*)d_in[7];
    const float* w3   = (const float*)d_in[8];
    const float* b3   = (const float*)d_in[9];
    const float* bias = (const float*)d_in[10];

    bf16_t* gbf  = (bf16_t*)d_ws;                               // 4,259,840 B
    bf16_t* xrev = (bf16_t*)((char*)d_ws + (size_t)COUT_*CIN_*GD*2); // 4,521,984 B

    ck_gen_taps<<<dim3((K_ + 255)/256), dim3(256), 0, stream>>>(
        rel, w1, b1, om1, w2, b2, om2, w3, b3, gbf);

    ck_rev_x<<<dim3(B_*CIN_), dim3(256), 0, stream>>>(x, xrev);

    ck_conv_wmma<<<dim3(T_/TBLK, B_), dim3(256), 0, stream>>>(
        xrev, gbf, bias, (float*)d_out);
}
